// MoeFF_35416300323104
// MI455X (gfx1250) — matrix operand order repaired and checked by reading, not compiled, not run on hardware
//
#include <hip/hip_runtime.h>
#include <hip/hip_bf16.h>

// ---------------------------------------------------------------------------
// MoE FF (SwiGLU, top-2 of 8 experts), B*S=4096 tokens, D=1024, H=4096.
// Route first, then per-expert bf16 WMMA GEMMs (v_wmma_f32_16x16x32_bf16),
// double-buffered LDS, async global->LDS copies for bf16 activations.
// ---------------------------------------------------------------------------

#define NTOK 4096   // B*S tokens
#define DD   1024   // model dim
#define HH   4096   // hidden dim
#define EE   8      // experts

#define BM 128      // token rows per tile
#define BN 64       // output cols per tile
#define BK 32       // k per step (WMMA K for bf16)
#define LDT 40      // padded LDS k-stride (elements); rows stay 16B aligned

typedef __attribute__((ext_vector_type(16))) __bf16 v16bf;
typedef __attribute__((ext_vector_type(8)))  __bf16 v8bf;
typedef __attribute__((ext_vector_type(2)))  __bf16 v2bf;
typedef __attribute__((ext_vector_type(8)))  float  v8f;
typedef __attribute__((ext_vector_type(4)))  int    v4i;

typedef __attribute__((address_space(1))) v4i* g4p;   // global int4*
typedef __attribute__((address_space(3))) v4i* l4p;   // LDS int4*

union FragBF { v16bf v; v8bf h[2]; };

#if defined(__has_builtin)
#if __has_builtin(__builtin_amdgcn_global_load_async_to_lds_b128)
#define HAVE_ASYNC_LDS 1
#endif
#if __has_builtin(__builtin_amdgcn_s_wait_asynccnt)
#define HAVE_WAIT_ASYNC 1
#endif
#if __has_builtin(__builtin_amdgcn_cvt_pk_bf16_f32)
#define HAVE_PK_BF16 1
#endif
#endif

// native bf16 convert (v_cvt*bf16_f32 on gfx1250)
__device__ __forceinline__ unsigned short f2bf(float f) {
    union { __bf16 b; unsigned short u; } v;
    v.b = (__bf16)f;
    return v.u;
}
// pack two f32 -> packed bf16 pair in one v_cvt_pk_bf16_f32 when available
__device__ __forceinline__ unsigned pack2bf(float lo, float hi) {
#if defined(HAVE_PK_BF16)
    union { v2bf v; unsigned u; } c;
    c.v = __builtin_amdgcn_cvt_pk_bf16_f32(lo, hi);
    return c.u;
#else
    return (unsigned)f2bf(lo) | ((unsigned)f2bf(hi) << 16);
#endif
}

// 16B global -> LDS copy; async (ASYNCcnt) when the builtin exists
__device__ __forceinline__ void copy16_g2l(unsigned short* lds, const unsigned short* g) {
#if defined(HAVE_ASYNC_LDS)
    v4i* gp = (v4i*)g;     // C-style cast drops const
    v4i* lp = (v4i*)lds;
    __builtin_amdgcn_global_load_async_to_lds_b128((g4p)gp, (l4p)lp, 0, 0);
#else
    *(uint4*)lds = *(const uint4*)g;
#endif
}
__device__ __forceinline__ void wait_async_all() {
#if defined(HAVE_ASYNC_LDS)
#if defined(HAVE_WAIT_ASYNC)
    __builtin_amdgcn_s_wait_asynccnt(0);
#else
    asm volatile("s_wait_asynccnt 0x0" ::: "memory");
#endif
#endif
}

// A-fragment: 16x32 bf16, M = lane&15; lanes<16 hold K 0..7 & 16..23,
// lanes>=16 hold K 8..15 & 24..31 (ISA 7.12.2). Two 16B LDS loads.
__device__ __forceinline__ v16bf load_a_frag(const unsigned short* sm, int mbase) {
    int lane = threadIdx.x & 31;
    int m = lane & 15, half = lane >> 4;
    const unsigned short* row = sm + (mbase + m) * LDT;
    FragBF f;
    f.h[0] = *(const v8bf*)(row + half * 8);
    f.h[1] = *(const v8bf*)(row + 16 + half * 8);
    return f.v;
}

// B-fragment: 32x16 bf16 staged transposed (Wt[n][k], k contiguous).
__device__ __forceinline__ v16bf load_b_frag(const unsigned short* sm, int nbase) {
    int lane = threadIdx.x & 31;
    int n = lane & 15, half = lane >> 4;
    const unsigned short* p = sm + (nbase + n) * LDT + half * 8;
    FragBF f;
    f.h[0] = *(const v8bf*)(p);
    f.h[1] = *(const v8bf*)(p + 16);
    return f.v;
}

// ---------------------------------------------------------------------------
// Kernel 1: zero output + expert counters
// ---------------------------------------------------------------------------
__global__ __launch_bounds__(256) void moe_init_kernel(float* __restrict__ out,
                                                       int* __restrict__ cnt) {
    int i0 = blockIdx.x * blockDim.x + threadIdx.x;
    int stride = gridDim.x * blockDim.x;
    const int total = NTOK * DD;
    for (int i = i0; i < total; i += stride) out[i] = 0.0f;
    if (blockIdx.x == 0 && threadIdx.x < EE) cnt[threadIdx.x] = 0;
}

// ---------------------------------------------------------------------------
// Kernel 2: x fp32 -> bf16
// ---------------------------------------------------------------------------
__global__ __launch_bounds__(256) void xconv_kernel(const float* __restrict__ x,
                                                    unsigned short* __restrict__ xbf) {
    int i0 = blockIdx.x * blockDim.x + threadIdx.x;
    int stride = gridDim.x * blockDim.x;
    const int total = NTOK * DD;
    for (int i = i0; i < total; i += stride) xbf[i] = f2bf(x[i]);
}

// ---------------------------------------------------------------------------
// Kernel 3: gate = softmax(x@Wg + bg), top-2, renormalize, build expert lists
// One wave32 per token (8 waves / block of 256).
// ---------------------------------------------------------------------------
__global__ __launch_bounds__(256) void gate_kernel(const float* __restrict__ x,
                                                   const float* __restrict__ Wg,   // [D,E]
                                                   const float* __restrict__ bg,   // [E]
                                                   int*   __restrict__ cnt,        // [E]
                                                   int*   __restrict__ tokl,       // [E,NTOK]
                                                   float* __restrict__ wgtl) {     // [E,NTOK]
    int wave = threadIdx.x >> 5;
    int lane = threadIdx.x & 31;
    int t = blockIdx.x * 8 + wave;

    float acc[EE];
#pragma unroll
    for (int e = 0; e < EE; ++e) acc[e] = 0.0f;

    const float* xp = x + (size_t)t * DD;
    for (int d = lane; d < DD; d += 32) {
        float xv = xp[d];
        const float* wr = Wg + (size_t)d * EE;
#pragma unroll
        for (int e = 0; e < EE; ++e) acc[e] += xv * wr[e];
    }
#pragma unroll
    for (int e = 0; e < EE; ++e) {
#pragma unroll
        for (int off = 16; off > 0; off >>= 1)
            acc[e] += __shfl_xor(acc[e], off, 32);
    }
    if (lane == 0) {
        float l[EE], m = -1e30f;
#pragma unroll
        for (int e = 0; e < EE; ++e) { l[e] = acc[e] + bg[e]; m = fmaxf(m, l[e]); }
        float p[EE];
#pragma unroll
        for (int e = 0; e < EE; ++e) p[e] = __expf(l[e] - m);
        int i0 = 0;
#pragma unroll
        for (int e = 1; e < EE; ++e) if (p[e] > p[i0]) i0 = e;
        int i1 = (i0 == 0) ? 1 : 0;
#pragma unroll
        for (int e = 0; e < EE; ++e) if (e != i0 && p[e] > p[i1]) i1 = e;
        float wsum = p[i0] + p[i1];
        float rn = __builtin_amdgcn_rcpf(wsum);
        int s0 = atomicAdd(&cnt[i0], 1);
        tokl[i0 * NTOK + s0] = t; wgtl[i0 * NTOK + s0] = p[i0] * rn;
        int s1 = atomicAdd(&cnt[i1], 1);
        tokl[i1 * NTOK + s1] = t; wgtl[i1 * NTOK + s1] = p[i1] * rn;
    }
}

// ---------------------------------------------------------------------------
// Kernel 4: per-expert up-projection  G = silu(X@Wa+ba) * (X@W1+b1)  [bf16]
// grid = (HH/BN, NTOK/BM, EE); 256 threads = 8 waves (2 in M x 4 in N).
// Double-buffered LDS; X tile via async-to-LDS; W1/Wa prefetched to regs.
// ---------------------------------------------------------------------------
__global__ __launch_bounds__(256) void moe_up_kernel(
        const unsigned short* __restrict__ xbf,   // [NTOK, DD] bf16
        const float* __restrict__ W1,             // [E, D, H]
        const float* __restrict__ b1,             // [E, H]
        const float* __restrict__ Wa,             // [E, D, H]
        const float* __restrict__ ba,             // [E, H]
        const int*   __restrict__ cnt,
        const int*   __restrict__ tokl,
        unsigned short* __restrict__ G) {         // [E*NTOK, HH] bf16
    __shared__ __align__(16) unsigned short smX [2][BM * LDT];
    __shared__ __align__(16) unsigned short smW1[2][BN * LDT];
    __shared__ __align__(16) unsigned short smWa[2][BN * LDT];
    __shared__ int smTok[BM];

    const int e     = blockIdx.z;
    const int mtile = blockIdx.y;
    const int h0    = blockIdx.x * BN;
    const int ce    = cnt[e];
    const int m0    = mtile * BM;
    if (m0 >= ce) return;                          // uniform per-block exit

    const int tid  = threadIdx.x;
    const int wave = tid >> 5;
    const int lane = tid & 31;
    const int wm   = wave >> 2;                    // 0..1  (64-row slab)
    const int wn   = wave & 3;                     // 0..3  (16-col slab)

    // staging thread mapping
    const int xr  = tid >> 1;                      // X row 0..127
    const int xkh = (tid & 1) * 16;                // X k-half
    const int wnc = tid & 63;                      // W col n
    const int wkg = (tid >> 6) * 8;                // W k-group base (0,8,16,24)

    if (tid < BM) {
        int slot = m0 + tid;
        smTok[tid] = tokl[e * NTOK + (slot < ce ? slot : m0)];
    }
    __syncthreads();

    const float* W1e = W1 + (size_t)e * DD * HH;
    const float* Wae = Wa + (size_t)e * DD * HH;
    const unsigned short* xrow = xbf + (size_t)smTok[xr] * DD + xkh;

    v8f zero{};
    v8f accU[4], accA[4];
#pragma unroll
    for (int i = 0; i < 4; ++i) { accU[i] = zero; accA[i] = zero; }

    float rw1[8], rwa[8];

    // ---- prologue: stage k-step 0 into buffer 0
    copy16_g2l(&smX[0][xr * LDT + xkh],     xrow);
    copy16_g2l(&smX[0][xr * LDT + xkh + 8], xrow + 8);
#pragma unroll
    for (int j = 0; j < 8; ++j) {
        size_t gi = (size_t)(wkg + j) * HH + h0 + wnc;
        rw1[j] = W1e[gi];
        rwa[j] = Wae[gi];
    }
    {
        uint4 u1, ua;
        u1.x = pack2bf(rw1[0], rw1[1]); u1.y = pack2bf(rw1[2], rw1[3]);
        u1.z = pack2bf(rw1[4], rw1[5]); u1.w = pack2bf(rw1[6], rw1[7]);
        ua.x = pack2bf(rwa[0], rwa[1]); ua.y = pack2bf(rwa[2], rwa[3]);
        ua.z = pack2bf(rwa[4], rwa[5]); ua.w = pack2bf(rwa[6], rwa[7]);
        *(uint4*)&smW1[0][wnc * LDT + wkg] = u1;
        *(uint4*)&smWa[0][wnc * LDT + wkg] = ua;
    }
    wait_async_all();

    const int NK = DD / BK;                        // 32 k-steps
    for (int kt = 0; kt < NK; ++kt) {
        const int buf = kt & 1;
        __syncthreads();                           // publish buf, free buf^1

        if (kt + 1 < NK) {                         // issue next-tile loads early
            const unsigned short* xs = xrow + (kt + 1) * BK;
            copy16_g2l(&smX[buf ^ 1][xr * LDT + xkh],     xs);
            copy16_g2l(&smX[buf ^ 1][xr * LDT + xkh + 8], xs + 8);
#pragma unroll
            for (int j = 0; j < 8; ++j) {
                size_t gi = (size_t)((kt + 1) * BK + wkg + j) * HH + h0 + wnc;
                rw1[j] = W1e[gi];
                rwa[j] = Wae[gi];
            }
        }

        // ---- compute on buf (overlaps with in-flight global loads)
        v16bf bU = load_b_frag(smW1[buf], wn * 16);
        v16bf bA = load_b_frag(smWa[buf], wn * 16);
#pragma unroll
        for (int i = 0; i < 4; ++i) {
            v16bf af = load_a_frag(smX[buf], wm * 64 + i * 16);
            accU[i] = __builtin_amdgcn_wmma_f32_16x16x32_bf16(
                          false, af, false, bU, (short)0, accU[i], false, false);
            accA[i] = __builtin_amdgcn_wmma_f32_16x16x32_bf16(
                          false, af, false, bA, (short)0, accA[i], false, false);
        }

        if (kt + 1 < NK) {                         // convert + store next weights
            uint4 u1, ua;
            u1.x = pack2bf(rw1[0], rw1[1]); u1.y = pack2bf(rw1[2], rw1[3]);
            u1.z = pack2bf(rw1[4], rw1[5]); u1.w = pack2bf(rw1[6], rw1[7]);
            ua.x = pack2bf(rwa[0], rwa[1]); ua.y = pack2bf(rwa[2], rwa[3]);
            ua.z = pack2bf(rwa[4], rwa[5]); ua.w = pack2bf(rwa[6], rwa[7]);
            *(uint4*)&smW1[buf ^ 1][wnc * LDT + wkg] = u1;
            *(uint4*)&smWa[buf ^ 1][wnc * LDT + wkg] = ua;
            wait_async_all();                      // drain X async before publish
        }
    }

    // ---- epilogue: SiLU(A+ba) * (U+b1) -> G (bf16)
    int n = lane & 15, halfsel = lane >> 4;
    int h = h0 + wn * 16 + n;
    float bb1 = b1[e * HH + h];
    float bba = ba[e * HH + h];
#pragma unroll
    for (int i = 0; i < 4; ++i) {
#pragma unroll
        for (int r = 0; r < 8; ++r) {
            int m    = wm * 64 + i * 16 + halfsel * 8 + r;
            int slot = m0 + m;
            if (slot < ce) {
                float a = accA[i][r] + bba;
                float u = accU[i][r] + bb1;
                float s = a * __builtin_amdgcn_rcpf(1.0f + __expf(-a));  // silu
                G[((size_t)e * NTOK + slot) * HH + h] = f2bf(s * u);
            }
        }
    }
}

// ---------------------------------------------------------------------------
// Kernel 5: per-expert down-projection  Y = G@W2 + b2, out[token] += w * Y
// grid = (DD/BN, NTOK/BM, EE); same pipelined structure.
// ---------------------------------------------------------------------------
__global__ __launch_bounds__(256) void moe_down_kernel(
        const unsigned short* __restrict__ G,     // [E*NTOK, HH] bf16
        const float* __restrict__ W2,             // [E, H, D]
        const float* __restrict__ b2,             // [E, D]
        const int*   __restrict__ cnt,
        const int*   __restrict__ tokl,
        const float* __restrict__ wgtl,
        float* __restrict__ out) {                // [NTOK, DD]
    __shared__ __align__(16) unsigned short smG [2][BM * LDT];
    __shared__ __align__(16) unsigned short smW2[2][BN * LDT];
    __shared__ int   smTok[BM];
    __shared__ float smWgt[BM];

    const int e     = blockIdx.z;
    const int mtile = blockIdx.y;
    const int d0    = blockIdx.x * BN;
    const int ce    = cnt[e];
    const int m0    = mtile * BM;
    if (m0 >= ce) return;

    const int tid  = threadIdx.x;
    const int wave = tid >> 5;
    const int lane = tid & 31;
    const int wm   = wave >> 2;
    const int wn   = wave & 3;

    const int xr  = tid >> 1;
    const int xkh = (tid & 1) * 16;
    const int wnc = tid & 63;
    const int wkg = (tid >> 6) * 8;

    if (tid < BM) {
        int slot = m0 + tid;
        int cs   = (slot < ce) ? slot : m0;
        smTok[tid] = tokl[e * NTOK + cs];
        smWgt[tid] = wgtl[e * NTOK + cs];
    }
    __syncthreads();

    const float* W2e = W2 + (size_t)e * HH * DD;
    const unsigned short* grow =
        G + ((size_t)e * NTOK + ((m0 + xr < ce) ? (m0 + xr) : m0)) * HH + xkh;

    v8f zero{};
    v8f acc[4];
#pragma unroll
    for (int i = 0; i < 4; ++i) acc[i] = zero;

    float rw2[8];

    // ---- prologue
    copy16_g2l(&smG[0][xr * LDT + xkh],     grow);
    copy16_g2l(&smG[0][xr * LDT + xkh + 8], grow + 8);
#pragma unroll
    for (int j = 0; j < 8; ++j)
        rw2[j] = W2e[(size_t)(wkg + j) * DD + d0 + wnc];
    {
        uint4 u;
        u.x = pack2bf(rw2[0], rw2[1]); u.y = pack2bf(rw2[2], rw2[3]);
        u.z = pack2bf(rw2[4], rw2[5]); u.w = pack2bf(rw2[6], rw2[7]);
        *(uint4*)&smW2[0][wnc * LDT + wkg] = u;
    }
    wait_async_all();

    const int NK = HH / BK;                        // 128 k-steps
    for (int kt = 0; kt < NK; ++kt) {
        const int buf = kt & 1;
        __syncthreads();

        if (kt + 1 < NK) {
            const unsigned short* gs = grow + (kt + 1) * BK;
            copy16_g2l(&smG[buf ^ 1][xr * LDT + xkh],     gs);
            copy16_g2l(&smG[buf ^ 1][xr * LDT + xkh + 8], gs + 8);
#pragma unroll
            for (int j = 0; j < 8; ++j)
                rw2[j] = W2e[(size_t)((kt + 1) * BK + wkg + j) * DD + d0 + wnc];
        }

        v16bf bf = load_b_frag(smW2[buf], wn * 16);
#pragma unroll
        for (int i = 0; i < 4; ++i) {
            v16bf af = load_a_frag(smG[buf], wm * 64 + i * 16);
            acc[i] = __builtin_amdgcn_wmma_f32_16x16x32_bf16(
                         false, af, false, bf, (short)0, acc[i], false, false);
        }

        if (kt + 1 < NK) {
            uint4 u;
            u.x = pack2bf(rw2[0], rw2[1]); u.y = pack2bf(rw2[2], rw2[3]);
            u.z = pack2bf(rw2[4], rw2[5]); u.w = pack2bf(rw2[6], rw2[7]);
            *(uint4*)&smW2[buf ^ 1][wnc * LDT + wkg] = u;
            wait_async_all();
        }
    }

    int n = lane & 15, halfsel = lane >> 4;
    int d = d0 + wn * 16 + n;
    float bb = b2[e * DD + d];
#pragma unroll
    for (int i = 0; i < 4; ++i) {
#pragma unroll
        for (int r = 0; r < 8; ++r) {
            int m    = wm * 64 + i * 16 + halfsel * 8 + r;
            int slot = m0 + m;
            if (slot < ce) {
                float y = acc[i][r] + bb;
                atomicAdd(&out[(size_t)smTok[m] * DD + d], smWgt[m] * y);
            }
        }
    }
}

// ---------------------------------------------------------------------------
// Host launcher
// ---------------------------------------------------------------------------
extern "C" void kernel_launch(void* const* d_in, const int* in_sizes, int n_in,
                              void* d_out, int out_size, void* d_ws, size_t ws_size,
                              hipStream_t stream) {
    const float* x  = (const float*)d_in[0];
    const float* Wg = (const float*)d_in[1];
    const float* bg = (const float*)d_in[2];
    const float* W1 = (const float*)d_in[3];
    const float* b1 = (const float*)d_in[4];
    const float* Wa = (const float*)d_in[5];
    const float* ba = (const float*)d_in[6];
    const float* W2 = (const float*)d_in[7];
    const float* b2 = (const float*)d_in[8];
    float* out = (float*)d_out;

    // Workspace layout (256B-aligned offsets)
    char* ws = (char*)d_ws;
    int*            cnt  = (int*)(ws + 0);                       // E ints
    int*            tokl = (int*)(ws + 1024);                    // E*NTOK ints
    float*          wgtl = (float*)(ws + 1024 + EE * NTOK * 4);  // E*NTOK floats
    unsigned short* xbf  = (unsigned short*)(ws + 1024 + 2 * EE * NTOK * 4);
    unsigned short* G    = (unsigned short*)((char*)xbf + (size_t)NTOK * DD * 2);

    moe_init_kernel<<<4096, 256, 0, stream>>>(out, cnt);
    xconv_kernel<<<2048, 256, 0, stream>>>(x, xbf);
    gate_kernel<<<NTOK / 8, 256, 0, stream>>>(x, Wg, bg, cnt, tokl, wgtl);

    dim3 gUp(HH / BN, NTOK / BM, EE);     // (64, 32, 8)
    moe_up_kernel<<<gUp, 256, 0, stream>>>(xbf, W1, b1, Wa, ba, cnt, tokl, G);

    dim3 gDn(DD / BN, NTOK / BM, EE);     // (16, 32, 8)
    moe_down_kernel<<<gDn, 256, 0, stream>>>(G, W2, b2, cnt, tokl, wgtl, out);
}
